// FBSNN_82927228551393
// MI455X (gfx1250) — compile-verified, matrix operand order checked by reading, not run on hardware
//
#include <hip/hip_runtime.h>
#include <hip/hip_bf16.h>

typedef _Float16 h8   __attribute__((ext_vector_type(8)));
typedef _Float16 v16h __attribute__((ext_vector_type(16)));
typedef float    v8f  __attribute__((ext_vector_type(8)));

namespace fbsnn {

constexpr int   D      = 100;
constexpr int   MPATH  = 2048;
constexpr int   NSTEP  = 20;
constexpr int   NT     = 21;          // N+1 time points
constexpr int   HID    = 1024;
constexpr int   INPAD  = 128;         // 101 padded to 128 for WMMA K
constexpr float R_     = 0.05f;
constexpr float MU_    = 0.06f;
constexpr float SIG_   = 0.2f;
constexpr float RHO_   = 0.5f;
constexpr float KSTRK  = 1.0f;

constexpr int    ROWS_G = MPATH * NT;               // 43008
constexpr size_t XPATH  = (size_t)MPATH * NT * D;   // X_path elements
constexpr size_t YPATH  = (size_t)MPATH * NT;       // Y_path elements

// ---- workspace layout (bytes, chained, all 256B aligned) ----
constexpr size_t OFF_L    = 0;
constexpr size_t SZ_L     = ((size_t)D * D * 4 + 255) & ~(size_t)255;       // 40192
constexpr size_t OFF_G    = OFF_L + SZ_L;
constexpr size_t SZ_G     = (size_t)ROWS_G * D * 4;                          // 17203200
constexpr size_t OFF_W1   = OFF_G + SZ_G;
constexpr size_t SZ_W1    = (size_t)INPAD * HID * 2;                         // 262144
constexpr size_t OFF_W1T  = OFF_W1 + SZ_W1;
constexpr size_t OFF_W2   = OFF_W1T + SZ_W1;
constexpr size_t SZ_W     = (size_t)HID * HID * 2;                           // 2097152
constexpr size_t OFF_W2T  = OFF_W2 + SZ_W;
constexpr size_t OFF_W3   = OFF_W2T + SZ_W;
constexpr size_t OFF_W3T  = OFF_W3 + SZ_W;
constexpr size_t OFF_W4   = OFF_W3T + SZ_W;
constexpr size_t OFF_W4T  = OFF_W4 + SZ_W;
constexpr size_t OFF_Z    = OFF_W4T + SZ_W;
constexpr size_t SZ_Z     = (size_t)MPATH * INPAD * 2;                       // 524288
constexpr size_t OFF_H1   = OFF_Z + SZ_Z;
constexpr size_t SZ_H     = (size_t)MPATH * HID * 2;                         // 4194304
constexpr size_t OFF_H2   = OFF_H1 + SZ_H;
constexpr size_t OFF_H3   = OFF_H2 + SZ_H;
constexpr size_t OFF_H4   = OFF_H3 + SZ_H;
constexpr size_t OFF_GA0  = OFF_H4 + SZ_H;
constexpr size_t OFF_GA1  = OFF_GA0 + SZ_H;
constexpr size_t OFF_DUDX = OFF_GA1 + SZ_H;
constexpr size_t SZ_DUDX  = (size_t)MPATH * INPAD * 4;                       // 1048576
constexpr size_t OFF_S    = OFF_DUDX + SZ_DUDX;
constexpr size_t SZ_S     = (size_t)MPATH * NSTEP * 4;                       // 163840
// total ~54.6 MiB

} // namespace fbsnn
using namespace fbsnn;

// ---------------------------------------------------------------------------
// Setup kernels
// ---------------------------------------------------------------------------

__global__ void init_kernel(float* out) {
    if (threadIdx.x == 0 && blockIdx.x == 0) out[0] = 0.0f;  // loss accumulator
}

// Cholesky of COV = sigma^2 * (rho + (1-rho) I), 100x100, one workgroup in LDS.
__global__ void cholesky_kernel(float* __restrict__ L) {
    __shared__ float A[D * D];
    for (int i = threadIdx.x; i < D * D; i += blockDim.x) {
        int r = i / D, c = i % D;
        A[i] = SIG_ * SIG_ * (RHO_ + (r == c ? (1.0f - RHO_) : 0.0f));
    }
    __syncthreads();
    for (int j = 0; j < D; ++j) {
        if (threadIdx.x == 0) {
            float s = A[j * D + j];
            for (int k = 0; k < j; ++k) s -= A[j * D + k] * A[j * D + k];
            A[j * D + j] = sqrtf(s);
        }
        __syncthreads();
        float djj = A[j * D + j];
        for (int i = j + 1 + threadIdx.x; i < D; i += blockDim.x) {
            float s = A[i * D + j];
            for (int k = 0; k < j; ++k) s -= A[i * D + k] * A[j * D + k];
            A[i * D + j] = s / djj;
        }
        __syncthreads();
    }
    for (int i = threadIdx.x; i < D * D; i += blockDim.x) {
        int r = i / D, c = i % D;
        L[i] = (c <= r) ? A[i] : 0.0f;
    }
}

// G[r,j] = sum_d Wpath[r,d] * L[j,d]   (i.e. W @ L^T), r over all M*(N+1) rows.
__global__ void g_kernel(const float* __restrict__ Wpath, const float* __restrict__ L,
                         float* __restrict__ G) {
    int idx = blockIdx.x * blockDim.x + threadIdx.x;
    if (idx >= ROWS_G * D) return;
    int r = idx / D, j = idx % D;
    const float* wrow = Wpath + (size_t)r * D;
    const float* lrow = L + (size_t)j * D;
    float s = 0.0f;
    for (int d = 0; d <= j; ++d) s += wrow[d] * lrow[d];  // L is lower-triangular
    G[idx] = s;
}

// Closed-form X path straight into d_out: X[m,n,d] = Xi[d]*exp((mu-.5*sig^2)*t + G)
__global__ void x_kernel(const float* __restrict__ t, const float* __restrict__ Xi,
                         const float* __restrict__ G, float* __restrict__ out) {
    int idx = blockIdx.x * blockDim.x + threadIdx.x;
    if (idx >= ROWS_G * D) return;
    int r = idx / D, d = idx % D;
    float drift = (MU_ - 0.5f * SIG_ * SIG_) * t[r];
    out[1 + (size_t)idx] = Xi[d] * expf(drift + G[idx]);
}

// Pack fp32 weight (inRows x HID) to f16, zero-padded to inPad rows, + transpose.
__global__ void pack_kernel(const float* __restrict__ src, int inRows, int inPad,
                            _Float16* __restrict__ w, _Float16* __restrict__ wt) {
    int idx = blockIdx.x * blockDim.x + threadIdx.x;
    if (idx >= inPad * HID) return;
    int r = idx / HID, c = idx % HID;
    float v = (r < inRows) ? src[(size_t)r * HID + c] : 0.0f;
    w[idx] = (_Float16)v;
    wt[(size_t)c * inPad + r] = (_Float16)v;
}

// Build Z16 = [t_n | X_n | 0-pad] in f16 for time point n (X read from d_out).
__global__ void z_kernel(const float* __restrict__ t, const float* __restrict__ out,
                         _Float16* __restrict__ Z, int n) {
    int idx = blockIdx.x * blockDim.x + threadIdx.x;
    if (idx >= MPATH * INPAD) return;
    int m = idx / INPAD, c = idx % INPAD;
    float v;
    if (c == 0)       v = t[(size_t)m * NT + n];
    else if (c <= D)  v = out[1 + ((size_t)m * NT + n) * D + (c - 1)];
    else              v = 0.0f;
    Z[idx] = (_Float16)v;
}

// ---------------------------------------------------------------------------
// WMMA GEMM: C = A(f16, MxK) @ B(KxN) using Bt (NxK f16, row-major).
// Block = 8 waves; wave computes a 16x64 strip (A-frag reused over 4 N-tiles).
// B tile (64 cols x 32 k = 4KB/stage) is staged global->LDS with the CDNA5
// async copy path (ASYNCcnt), double-buffered, so the 8 waves share one copy
// instead of 8 redundant vector-load streams; fragments come from ds_load.
// MODE 0: h = tanh(acc + bias)       -> out16 (forward hidden layers)
// MODE 1: g = acc * (1 - h^2)        -> out16 (backward, h from Hact f16)
// MODE 2: acc                         -> out32 (backward-to-input)
// ---------------------------------------------------------------------------
template <int MODE>
__global__ __launch_bounds__(256) void wmma_gemm(
    const _Float16* __restrict__ A, int lda,
    const _Float16* __restrict__ Bt, int ldb,
    const float* __restrict__ bias,
    const _Float16* __restrict__ Hact, int ldh,
    _Float16* __restrict__ out16, float* __restrict__ out32, int ldo, int K) {
    constexpr int ROWP = 40;                    // padded LDS row stride (halves, 80B)
    __shared__ _Float16 tb[2][64 * ROWP];       // double-buffered B tile (10KB)

    const int tid   = threadIdx.x;
    const int lane  = tid & 31;
    const int wave  = tid >> 5;
    const int mBase = blockIdx.x * 128 + wave * 16;
    const int nBase = blockIdx.y * 64;
    const int r     = lane & 15;
    const int hi    = lane >> 4;  // K-half select per the 16-bit A 16x32 layout

    const _Float16* aRow = A + (size_t)(mBase + r) * lda;

    // Cooperative async B-tile fetch: 256 threads x 16B = 64 rows x 32 halves.
    const int bc = tid >> 2;                    // B-tile row (output column)
    const int bq = tid & 3;                     // 16B chunk within row
    const _Float16* bSrc = Bt + (size_t)(nBase + bc) * ldb + bq * 8;
    const unsigned  ldsRow = (unsigned)(bc * ROWP + bq * 8);

    v8f acc[4];
#pragma unroll
    for (int j = 0; j < 4; ++j)
        acc[j] = v8f{0.f, 0.f, 0.f, 0.f, 0.f, 0.f, 0.f, 0.f};

    const int stages = K >> 5;

    // Prologue: stage 0 -> buffer 0 (ASYNCcnt-tracked global->LDS copy).
    {
        unsigned dst = (unsigned)(size_t)&tb[0][ldsRow];
        asm volatile("global_load_async_to_lds_b128 %0, %1, off"
                     :: "v"(dst), "v"(bSrc) : "memory");
    }

    for (int s = 0; s < stages; ++s) {
        asm volatile("s_wait_asynccnt 0x0" ::: "memory");  // own copy for stage s done
        __syncthreads();  // all copies landed; all reads of the recycled buffer done
        if (s + 1 < stages) {
            unsigned dst = (unsigned)(size_t)&tb[(s + 1) & 1][ldsRow];
            asm volatile("global_load_async_to_lds_b128 %0, %1, off"
                         :: "v"(dst), "v"(bSrc + (size_t)(s + 1) * 32) : "memory");
        }
        const int k0 = (s << 5) + hi * 8;
        // A fragment: lanes 0-15 hold row r with K {kb..kb+7, kb+16..23},
        // lanes 16-31 hold K {kb+8..15, kb+24..31}  -> two 16B global loads.
        h8 alo = *(const h8*)(aRow + k0);
        h8 ahi = *(const h8*)(aRow + k0 + 16);
        v16h af;
#pragma unroll
        for (int i = 0; i < 8; ++i) { af[i] = alo[i]; af[i + 8] = ahi[i]; }
#pragma unroll
        for (int j = 0; j < 4; ++j) {
            const _Float16* bp = &tb[s & 1][(j * 16 + r) * ROWP + hi * 8];
            h8 blo = *(const h8*)bp;          // ds_load_b128
            h8 bhi = *(const h8*)(bp + 16);   // ds_load_b128
            v16h bf;
#pragma unroll
            for (int i = 0; i < 8; ++i) { bf[i] = blo[i]; bf[i + 8] = bhi[i]; }
            acc[j] = __builtin_amdgcn_wmma_f32_16x16x32_f16(
                false, af, false, bf, (short)0, acc[j], false, false);
        }
    }

    // Epilogue: 16x16 f32 C layout — VGPR v: lanes 0-15 row v, lanes 16-31 row v+8.
#pragma unroll
    for (int j = 0; j < 4; ++j) {
        const int col = nBase + j * 16 + (lane & 15);
        float b = 0.0f;
        if constexpr (MODE == 0) b = bias[col];
#pragma unroll
        for (int v = 0; v < 8; ++v) {
            const int row = mBase + ((lane < 16) ? v : v + 8);
            float val = acc[j][v] + b;
            const size_t idx = (size_t)row * ldo + col;
            if constexpr (MODE == 0) {
                out16[idx] = (_Float16)tanhf(val);
            } else if constexpr (MODE == 1) {
                float h = (float)Hact[(size_t)row * ldh + col];
                out16[idx] = (_Float16)(val * (1.0f - h * h));
            } else {
                out32[idx] = val;
            }
        }
    }
}

// ---------------------------------------------------------------------------
// Small per-timepoint kernels
// ---------------------------------------------------------------------------

// u[m] = h4[m,:] . W5 + b5  -> Y_path[m, n]
__global__ void u_kernel(const _Float16* __restrict__ h4, const float* __restrict__ W5,
                         const float* __restrict__ b5, float* __restrict__ yout, int n) {
    __shared__ float red[256];
    const int m = blockIdx.x;
    float s = 0.0f;
    for (int k = threadIdx.x; k < HID; k += 256)
        s += (float)h4[(size_t)m * HID + k] * W5[k];
    red[threadIdx.x] = s;
    __syncthreads();
    for (int off = 128; off > 0; off >>= 1) {
        if ((int)threadIdx.x < off) red[threadIdx.x] += red[threadIdx.x + off];
        __syncthreads();
    }
    if (threadIdx.x == 0) yout[(size_t)m * NT + n] = red[0] + b5[0];
}

// Backward seed: ga4 = W5 (.) (1 - h4^2), f16.
__global__ void seed_kernel(const _Float16* __restrict__ h4, const float* __restrict__ W5,
                            _Float16* __restrict__ ga) {
    int idx = blockIdx.x * blockDim.x + threadIdx.x;
    if (idx >= MPATH * HID) return;
    float h = (float)h4[idx];
    ga[idx] = (_Float16)(W5[idx % HID] * (1.0f - h * h));
}

// S[m,n] = DuDx_n[m,:] . (G[m,n+1,:]-G[m,n,:])  (LdW of step n+1)
__global__ void s_kernel(const float* __restrict__ dudx, const float* __restrict__ G,
                         float* __restrict__ S, int n) {
    int m = blockIdx.x * blockDim.x + threadIdx.x;
    if (m >= MPATH) return;
    const float* g0 = G + ((size_t)m * NT + n) * D;
    const float* g1 = g0 + D;
    const float* du = dudx + (size_t)m * INPAD + 1;  // skip t-column
    float s = 0.0f;
    for (int d = 0; d < D; ++d) s += du[d] * (g1[d] - g0[d]);
    S[(size_t)m * NSTEP + n] = s;
}

// Replays the scan's Y_tilde recursion + terminal payoff; atomicAdd into out[0].
__global__ void loss_kernel(const float* __restrict__ t, const float* __restrict__ S,
                            float* __restrict__ out) {
    int m = blockIdx.x * blockDim.x + threadIdx.x;
    if (m >= MPATH) return;
    const float* Yp = out + 1 + XPATH;
    float acc = 0.0f;
    for (int n = 1; n <= NSTEP; ++n) {
        float dt = t[(size_t)m * NT + n] - t[(size_t)m * NT + n - 1];
        float Y0 = Yp[(size_t)m * NT + n - 1];
        float Y1 = Yp[(size_t)m * NT + n];
        float til = Y0 + R_ * Y0 * dt + Y0 * S[(size_t)m * NSTEP + (n - 1)];
        float e = Y1 - til;
        acc += e * e;
    }
    const float* xN = out + 1 + ((size_t)m * NT + NSTEP) * D;
    float xm = 0.0f;
    for (int d = 0; d < D; ++d) xm += xN[d];
    float g = fmaxf(xm * (1.0f / D) - KSTRK, 0.0f);
    float e = Yp[(size_t)m * NT + NSTEP] - g;
    acc += e * e;
    atomicAdd(out, acc);
    if (m == 0) out[1 + XPATH + YPATH] = Yp[0];  // Y_path[0,0,0]
}

// ---------------------------------------------------------------------------
extern "C" void kernel_launch(void* const* d_in, const int* in_sizes, int n_in,
                              void* d_out, int out_size, void* d_ws, size_t ws_size,
                              hipStream_t stream) {
    (void)in_sizes; (void)n_in; (void)out_size; (void)ws_size;
    const float* t  = (const float*)d_in[0];   // (M, N+1, 1)
    const float* Wp = (const float*)d_in[1];   // (M, N+1, D)
    const float* Xi = (const float*)d_in[2];   // (D,)
    const float* Wl[5] = {(const float*)d_in[3], (const float*)d_in[5],
                          (const float*)d_in[7], (const float*)d_in[9],
                          (const float*)d_in[11]};
    const float* bl[5] = {(const float*)d_in[4], (const float*)d_in[6],
                          (const float*)d_in[8], (const float*)d_in[10],
                          (const float*)d_in[12]};
    float* out = (float*)d_out;
    char*  ws  = (char*)d_ws;

    float*     Lf   = (float*)(ws + OFF_L);
    float*     Gf   = (float*)(ws + OFF_G);
    _Float16*  W1   = (_Float16*)(ws + OFF_W1);
    _Float16*  W1T  = (_Float16*)(ws + OFF_W1T);
    _Float16*  W2   = (_Float16*)(ws + OFF_W2);
    _Float16*  W2T  = (_Float16*)(ws + OFF_W2T);
    _Float16*  W3   = (_Float16*)(ws + OFF_W3);
    _Float16*  W3T  = (_Float16*)(ws + OFF_W3T);
    _Float16*  W4   = (_Float16*)(ws + OFF_W4);
    _Float16*  W4T  = (_Float16*)(ws + OFF_W4T);
    _Float16*  Z16  = (_Float16*)(ws + OFF_Z);
    _Float16*  H1   = (_Float16*)(ws + OFF_H1);
    _Float16*  H2   = (_Float16*)(ws + OFF_H2);
    _Float16*  H3   = (_Float16*)(ws + OFF_H3);
    _Float16*  H4   = (_Float16*)(ws + OFF_H4);
    _Float16*  GA0  = (_Float16*)(ws + OFF_GA0);
    _Float16*  GA1  = (_Float16*)(ws + OFF_GA1);
    float*     DUDX = (float*)(ws + OFF_DUDX);
    float*     Sf   = (float*)(ws + OFF_S);

    init_kernel<<<1, 32, 0, stream>>>(out);
    cholesky_kernel<<<1, 128, 0, stream>>>(Lf);

    const int gElems = ROWS_G * D;
    g_kernel<<<(gElems + 255) / 256, 256, 0, stream>>>(Wp, Lf, Gf);
    x_kernel<<<(gElems + 255) / 256, 256, 0, stream>>>(t, Xi, Gf, out);

    pack_kernel<<<(INPAD * HID + 255) / 256, 256, 0, stream>>>(Wl[0], D + 1, INPAD, W1, W1T);
    pack_kernel<<<(HID * HID) / 256, 256, 0, stream>>>(Wl[1], HID, HID, W2, W2T);
    pack_kernel<<<(HID * HID) / 256, 256, 0, stream>>>(Wl[2], HID, HID, W3, W3T);
    pack_kernel<<<(HID * HID) / 256, 256, 0, stream>>>(Wl[3], HID, HID, W4, W4T);

    const dim3 blk(256);
    const dim3 grdFull(MPATH / 128, HID / 64);   // 16 x 16
    const dim3 grdIn(MPATH / 128, INPAD / 64);   // 16 x 2
    float* Ypath = out + 1 + XPATH;

    for (int n = 0; n < NT; ++n) {
        z_kernel<<<(MPATH * INPAD + 255) / 256, 256, 0, stream>>>(t, out, Z16, n);
        // Forward: h = tanh(prev @ W + b)
        wmma_gemm<0><<<grdFull, blk, 0, stream>>>(Z16, INPAD, W1T, INPAD, bl[0],
                                                  nullptr, 0, H1, nullptr, HID, INPAD);
        wmma_gemm<0><<<grdFull, blk, 0, stream>>>(H1, HID, W2T, HID, bl[1],
                                                  nullptr, 0, H2, nullptr, HID, HID);
        wmma_gemm<0><<<grdFull, blk, 0, stream>>>(H2, HID, W3T, HID, bl[2],
                                                  nullptr, 0, H3, nullptr, HID, HID);
        wmma_gemm<0><<<grdFull, blk, 0, stream>>>(H3, HID, W4T, HID, bl[3],
                                                  nullptr, 0, H4, nullptr, HID, HID);
        u_kernel<<<MPATH, 256, 0, stream>>>(H4, Wl[4], bl[4], Ypath, n);
        // Backward to input gradient (DuDx)
        seed_kernel<<<(MPATH * HID + 255) / 256, 256, 0, stream>>>(H4, Wl[4], GA0);
        wmma_gemm<1><<<grdFull, blk, 0, stream>>>(GA0, HID, W4, HID, nullptr,
                                                  H3, HID, GA1, nullptr, HID, HID);
        wmma_gemm<1><<<grdFull, blk, 0, stream>>>(GA1, HID, W3, HID, nullptr,
                                                  H2, HID, GA0, nullptr, HID, HID);
        wmma_gemm<1><<<grdFull, blk, 0, stream>>>(GA0, HID, W2, HID, nullptr,
                                                  H1, HID, GA1, nullptr, HID, HID);
        wmma_gemm<2><<<grdIn, blk, 0, stream>>>(GA1, HID, W1, HID, nullptr,
                                                nullptr, 0, nullptr, DUDX, INPAD, HID);
        if (n < NSTEP)
            s_kernel<<<(MPATH + 255) / 256, 256, 0, stream>>>(DUDX, Gf, Sf, n);
    }

    loss_kernel<<<(MPATH + 255) / 256, 256, 0, stream>>>(t, Sf, out);
}
